// PointNet2Sampler_11433202942131
// MI455X (gfx1250) — matrix operand order repaired and checked by reading, not compiled, not run on hardware
//
#include <hip/hip_runtime.h>
#include <hip/hip_bf16.h>

typedef __bf16 bf16_t;
typedef __bf16 v16bf __attribute__((ext_vector_type(16)));
typedef float  v8f   __attribute__((ext_vector_type(8)));

#define Bdim 4
#define Ndim 8192
#define Mdim 2048
#define Cdim 64
#define K0   32
#define K1   64
#define KDIM 67          // 3 (rel pos) + 64 (features)
#define KPAD 96          // pad K to multiple of 32 for bf16 WMMA
#define OPC  64
#define OUTC 256
#define NCTR (Bdim * Mdim)   // 8192 centers

// f32 -> bf16 (round-to-nearest-even), bit-level so we don't rely on fptrunc codegen
static __device__ __forceinline__ unsigned short f2bfbits(float f) {
  unsigned u = __float_as_uint(f);
  unsigned r = u + 0x7FFFu + ((u >> 16) & 1u);
  return (unsigned short)(r >> 16);
}
static __device__ __forceinline__ bf16_t bits2bf(unsigned short s) {
  bf16_t b;
  __builtin_memcpy(&b, &s, 2);
  return b;
}
static __device__ __forceinline__ bf16_t f2bf(float f) { return bits2bf(f2bfbits(f)); }

// ---------------------------------------------------------------------------
// Kernel 1: per-center ring membership compaction (ordered stream compaction).
// One wave32 per center; ballot + prefix popcount; early exit when both rings
// are saturated (expected after ~256 of 8192 candidates for this data).
// ---------------------------------------------------------------------------
__global__ __launch_bounds__(256) void ring_select(
    const float* __restrict__ dist, int* __restrict__ idx0,
    int* __restrict__ idx1, int* __restrict__ cnt) {
  const int tid  = threadIdx.x;
  const int lane = tid & 31;
  const int gw   = blockIdx.x * 8 + (tid >> 5);          // center id, 0..NCTR-1
  const float* drow = dist + (size_t)gw * Ndim;

  int c0 = 0, c1 = 0;
  const unsigned pm = (lane == 0) ? 0u : (0xFFFFFFFFu >> (32 - lane));
  for (int j = 0; j < Ndim; j += 32) {
    if (c0 >= K0 && c1 >= K1) break;                     // wave-uniform early exit
    const float d = drow[j + lane];
    const bool h0 = (d >= 1.0f) && (d < 2.25f);          // ring0: [min^2=1, 1.5^2)
    const bool h1 = (d >= 2.25f) && (d < 9.0f);          // ring1: [1.5^2, 3^2)
    const unsigned m0 = (unsigned)__ballot(h0);
    const unsigned m1 = (unsigned)__ballot(h1);
    if (h0) { int s = c0 + __popc(m0 & pm); if (s < K0) idx0[(size_t)gw * K0 + s] = j + lane; }
    if (h1) { int s = c1 + __popc(m1 & pm); if (s < K1) idx1[(size_t)gw * K1 + s] = j + lane; }
    c0 += __popc(m0);
    c1 += __popc(m1);
  }
  if (lane == 0) {
    cnt[gw * 2 + 0] = c0 < K0 ? c0 : K0;
    cnt[gw * 2 + 1] = c1 < K1 ? c1 : K1;
  }
}

// ---------------------------------------------------------------------------
// Kernel 2: per-center gathered MLP + row max, bf16 WMMA.
// Block = 128 threads = 4 waves, one block per center.
// GEMM per ring: (k x 96) * (96 x 64), tiles 16x16x32, f32 accumulate.
// Wave w owns output columns [16w, 16w+16); loops M-tiles keeping running max.
// ---------------------------------------------------------------------------
__global__ __launch_bounds__(128) void ring_mlp_max(
    const float* __restrict__ positions, const float* __restrict__ features,
    const float* __restrict__ centers,
    const float* __restrict__ Wop0, const float* __restrict__ bop0,
    const float* __restrict__ Wop1, const float* __restrict__ bop1,
    const int* __restrict__ idx0, const int* __restrict__ idx1,
    const int* __restrict__ cnt, float* __restrict__ xbuf) {
  __shared__ unsigned short sA[K1 * KPAD];        // 64 x 96 bf16 (12 KB)
  __shared__ unsigned short sW[2][KPAD * OPC];    // 2 x 96 x 64 bf16 (24 KB)
  __shared__ float sCtr[4];

  const int tid = threadIdx.x;
  const int g = blockIdx.x;                       // center id
  const int b = g >> 11;                          // / Mdim (2048)
  const int m = g & (Mdim - 1);

  if (tid < 3) sCtr[tid] = centers[((size_t)b * Mdim + m) * 3 + tid];

  // Stage both ring weight matrices (K padded 67 -> 96 with zeros).
  for (int e = tid; e < KPAD * OPC; e += 128) {
    const int row = e >> 6, col = e & 63;
    const float w0 = (row < KDIM) ? Wop0[row * OPC + col] : 0.0f;
    const float w1 = (row < KDIM) ? Wop1[row * OPC + col] : 0.0f;
    sW[0][e] = f2bfbits(w0);
    sW[1][e] = f2bfbits(w1);
  }
  __syncthreads();

  const int lane = tid & 31;
  const int wave = tid >> 5;
  const int half = lane >> 4;
  const int lr   = lane & 15;
  const int n0   = wave * 16;

  for (int ring = 0; ring < 2; ++ring) {
    const int k = ring ? K1 : K0;
    const int count = cnt[g * 2 + ring];
    const int* idxp = ring ? (idx1 + (size_t)g * K1) : (idx0 + (size_t)g * K0);
    const float* bop = ring ? bop1 : bop0;

    // Stage A = [rel_pos | feat | 0-pad], rows >= count zeroed (match reference mask).
    for (int e = tid; e < k * KPAD; e += 128) {
      const int row = e / KPAD;
      const int col = e - row * KPAD;
      float v = 0.0f;
      if (row < count && col < KDIM) {
        const int id = idxp[row];
        if (col < 3)
          v = positions[((size_t)b * Ndim + id) * 3 + col] - sCtr[col];
        else
          v = features[((size_t)b * Ndim + id) * Cdim + (col - 3)];
      }
      sA[e] = f2bfbits(v);
    }
    __syncthreads();

    float colmax = 0.0f;                      // relu >= 0, invalid rows are 0
    const float bias = bop[n0 + lr];
    const int mtiles = k >> 4;                // 2 (ring0) or 4 (ring1)
    for (int mt = 0; mt < mtiles; ++mt) {
      v8f acc = {0.f, 0.f, 0.f, 0.f, 0.f, 0.f, 0.f, 0.f};
      const int arow = mt * 16 + lr;
#pragma unroll
      for (int kb = 0; kb < KPAD; kb += 32) {
        v16bf av, bv;
#pragma unroll
        for (int i = 0; i < 16; ++i) {
          // A k order used here (contiguous, same as B): K = i + 16*laneHalf
          const int ka = i + half * 16;
          av[i] = bits2bf(sA[arow * KPAD + kb + ka]);
          // B k order used here (matches A): K = i + 16*laneHalf, N = lane&15
          const int kr = kb + i + half * 16;
          bv[i] = bits2bf(sW[ring][kr * OPC + n0 + lr]);
        }
        acc = __builtin_amdgcn_wmma_f32_16x16x32_bf16(false, av, false, bv,
                                                      (short)0, acc, false, false);
      }
      // bias + relu + invalid-row zero + reduce max over the tile's 16 rows
#pragma unroll
      for (int j = 0; j < 8; ++j) {
        const int row = mt * 16 + j + half * 8;
        float v = fmaxf(acc[j] + bias, 0.0f);
        if (row >= count) v = 0.0f;
        colmax = fmaxf(colmax, v);
      }
    }
    // combine the two lane halves (rows 0-7 vs 8-15 of each tile)
    colmax = fmaxf(colmax, __shfl_xor(colmax, 16, 32));
    if (lane < 16) xbuf[(size_t)g * 128 + ring * 64 + n0 + lr] = colmax;
    __syncthreads();
  }
}

// ---------------------------------------------------------------------------
// Kernel 3: aggregation GEMM (8192 x 256) = relu((8192 x 128) * (128 x 256) + b)
// One wave per 16x16 output tile, 4 WMMA K-steps. 1024 blocks x 8 waves.
// ---------------------------------------------------------------------------
__global__ __launch_bounds__(256) void agg_gemm(
    const float* __restrict__ xbuf, const float* __restrict__ Wagg,
    const float* __restrict__ bagg, float* __restrict__ out) {
  const int tid  = threadIdx.x;
  const int lane = tid & 31;
  const int half = lane >> 4;
  const int lr   = lane & 15;
  const int wid  = blockIdx.x * 8 + (tid >> 5);    // tile id, 0..8191
  const int m0   = (wid >> 4) * 16;                // row tile (512 tiles)
  const int n0   = (wid & 15) * 16;                // col tile (16 tiles)

  const int arow = m0 + lr;
  const int col  = n0 + lr;
  v8f acc = {0.f, 0.f, 0.f, 0.f, 0.f, 0.f, 0.f, 0.f};
#pragma unroll
  for (int kb = 0; kb < 128; kb += 32) {
    v16bf av, bv;
#pragma unroll
    for (int i = 0; i < 16; ++i) {
      const int ka = i + half * 16;
      av[i] = f2bf(xbuf[(size_t)arow * 128 + kb + ka]);
      const int kr = kb + i + half * 16;
      bv[i] = f2bf(Wagg[(size_t)kr * OUTC + col]);
    }
    acc = __builtin_amdgcn_wmma_f32_16x16x32_bf16(false, av, false, bv,
                                                  (short)0, acc, false, false);
  }
  const float bias = bagg[col];
#pragma unroll
  for (int j = 0; j < 8; ++j) {
    const int row = m0 + j + half * 8;
    out[(size_t)row * OUTC + col] = fmaxf(acc[j] + bias, 0.0f);
  }
}

// ---------------------------------------------------------------------------
extern "C" void kernel_launch(void* const* d_in, const int* in_sizes, int n_in,
                              void* d_out, int out_size, void* d_ws, size_t ws_size,
                              hipStream_t stream) {
  const float* positions = (const float*)d_in[0];
  const float* features  = (const float*)d_in[1];
  const float* centers   = (const float*)d_in[2];
  const float* distances = (const float*)d_in[3];
  const float* Wop0      = (const float*)d_in[4];
  const float* bop0      = (const float*)d_in[5];
  const float* Wop1      = (const float*)d_in[6];
  const float* bop1      = (const float*)d_in[7];
  const float* Wagg      = (const float*)d_in[8];
  const float* bagg      = (const float*)d_in[9];
  float* out = (float*)d_out;

  // Workspace carve-up (~7.4 MB total)
  int* idx0 = (int*)d_ws;                               // NCTR * 32 ints
  int* idx1 = idx0 + (size_t)NCTR * K0;                 // NCTR * 64 ints
  int* cnt  = idx1 + (size_t)NCTR * K1;                 // NCTR * 2  ints
  float* xbuf = (float*)(cnt + (size_t)NCTR * 2);       // NCTR * 128 f32

  ring_select<<<NCTR / 8, 256, 0, stream>>>(distances, idx0, idx1, cnt);
  ring_mlp_max<<<NCTR, 128, 0, stream>>>(positions, features, centers,
                                         Wop0, bop0, Wop1, bop1,
                                         idx0, idx1, cnt, xbuf);
  agg_gemm<<<(NCTR / 16) * (OUTC / 16) / 8, 256, 0, stream>>>(xbuf, Wagg, bagg, out);
}
